// MultiHeadAttention_2525440770592
// MI455X (gfx1250) — matrix operand order repaired and checked by reading, not compiled, not run on hardware
//
#include <hip/hip_runtime.h>

typedef __bf16 bf16_t;
typedef __attribute__((ext_vector_type(16))) __bf16 v16bf;
typedef __attribute__((ext_vector_type(8)))  __bf16 v8bf;
typedef __attribute__((ext_vector_type(8)))  float  v8f;

#define BB 2
#define SS 2048
#define DD 768
#define HH 12
#define DK 64
#define BS (BB*SS)
#define SCLD 2068   // fp32 score pitch (floats); 8*SCLD % 64 != 0 -> conflict-free wmma stores
#define PLD  2064   // bf16 probability pitch (elements); 16B-aligned rows

__device__ __forceinline__ v8f wmma_bf16(v16bf a, v16bf b, v8f c) {
  // D = A(16x32 bf16) x B(32x16 bf16) + C(16x16 f32)
  return __builtin_amdgcn_wmma_f32_16x16x32_bf16(false, a, false, b, (short)0, c,
                                                 false, false);
}

__device__ __forceinline__ v16bf pack16(v8bf lo, v8bf hi) {
  v16bf a;
#pragma unroll
  for (int i = 0; i < 8; ++i) { a[i] = lo[i]; a[8 + i] = hi[i]; }
  return a;
}

// A fragment: 16x32 bf16, row-major source, ld = row stride (elements).
// Lane l: row m = l&15; K-range [kb, kb+16) contiguous, kb = (l&16)?16:0.
__device__ __forceinline__ v16bf frag_a(const bf16_t* base, int ld, int lane) {
  int m  = lane & 15;
  int kb = (lane & 16) ? 16 : 0;
  const bf16_t* row = base + (size_t)m * ld + kb;
  return pack16(*(const v8bf*)(row), *(const v8bf*)(row + 8));
}

// B fragment: B[k][n] = Src[n][k] with Src row-major (the "@ W^T" pattern).
// Lane l: col n = l&15, K-range half = (l&16)?16:0, 16 contiguous elements.
__device__ __forceinline__ v16bf frag_b_rows(const bf16_t* base, int ld, int lane) {
  const bf16_t* p = base + (size_t)(lane & 15) * ld + ((lane & 16) ? 16 : 0);
  return *(const v16bf*)p;
}

// ---------------------------------------------------------------- convert
__global__ void k_cvt(const float* __restrict__ src, bf16_t* __restrict__ dst, int n) {
  int i = blockIdx.x * blockDim.x + threadIdx.x;
  int stride = gridDim.x * blockDim.x;
  for (; i < n; i += stride) dst[i] = (bf16_t)src[i];
}

// ---------------------------------------------------------------- projection (split-head)
// out[m][n] = sum_k X[m][k]*W[n][k] + bias[n], written split-head bf16 [B,H,S,64]
__global__ void k_proj(const bf16_t* __restrict__ X, const bf16_t* __restrict__ W,
                       const float* __restrict__ bias, bf16_t* __restrict__ outH) {
  int lane = threadIdx.x & 31;
  int waveId = blockIdx.x * (blockDim.x >> 5) + (threadIdx.x >> 5);
  const int NT = DD / 16;                 // 48 col tiles
  int tm = waveId / NT, tn = waveId % NT;
  int m0 = tm * 16, n0 = tn * 16;
  v8f acc = {};
  for (int k0 = 0; k0 < DD; k0 += 32) {
    v16bf a = frag_a(X + (size_t)m0 * DD + k0, DD, lane);
    v16bf b = frag_b_rows(W + (size_t)n0 * DD + k0, DD, lane);
    acc = wmma_bf16(a, b, acc);
  }
  int n = n0 + (lane & 15);
  float bia = bias[n];
  int h = n >> 6, d = n & 63;
  int mbase = m0 + ((lane & 16) ? 8 : 0);
#pragma unroll
  for (int v = 0; v < 8; ++v) {
    int r = mbase + v;
    int batch = r >> 11, s = r & (SS - 1);
    outH[(((size_t)(batch * HH + h) * SS + s) * DK) + d] = (bf16_t)(acc[v] + bia);
  }
}

// ---------------------------------------------------------------- projection (transposed head)
// Same GEMM, but written as V^T: [B,H,DK,S] so the P@V B-fragments are contiguous.
__global__ void k_projT(const bf16_t* __restrict__ X, const bf16_t* __restrict__ W,
                        const float* __restrict__ bias, bf16_t* __restrict__ outT) {
  int lane = threadIdx.x & 31;
  int waveId = blockIdx.x * (blockDim.x >> 5) + (threadIdx.x >> 5);
  const int NT = DD / 16;
  int tm = waveId / NT, tn = waveId % NT;
  int m0 = tm * 16, n0 = tn * 16;
  v8f acc = {};
  for (int k0 = 0; k0 < DD; k0 += 32) {
    v16bf a = frag_a(X + (size_t)m0 * DD + k0, DD, lane);
    v16bf b = frag_b_rows(W + (size_t)n0 * DD + k0, DD, lane);
    acc = wmma_bf16(a, b, acc);
  }
  int n = n0 + (lane & 15);
  float bia = bias[n];
  int h = n >> 6, d = n & 63;
  int mbase = m0 + ((lane & 16) ? 8 : 0);
#pragma unroll
  for (int v = 0; v < 8; ++v) {
    int r = mbase + v;
    int batch = r >> 11, s = r & (SS - 1);
    outT[(((size_t)(batch * HH + h) * DK + d) * SS) + s] = (bf16_t)(acc[v] + bia);
  }
}

// ---------------------------------------------------------------- fused attention
// One workgroup = one (batch, head, 16-query tile). 4 waves, 128 threads.
// LDS: fp32 scores [16][SCLD] + reduction[128] + bf16 probabilities [16][PLD].
__global__ void k_attn(const bf16_t* __restrict__ qh, const bf16_t* __restrict__ kh,
                       const bf16_t* __restrict__ vt, float* __restrict__ attn,
                       bf16_t* __restrict__ ctx) {
  extern __shared__ float smem[];
  float*  sc  = smem;                       // 16*SCLD fp32
  float*  red = smem + 16 * SCLD;           // 128 fp32
  bf16_t* pb  = (bf16_t*)(red + 128);       // 16*PLD bf16

  int tid = threadIdx.x, lane = tid & 31, wave = tid >> 5;
  int wg = blockIdx.x;
  int qt = wg & 127;
  int bh = wg >> 7;
  int h = bh % HH, batch = bh / HH;
  int q0 = qt * 16;

  const bf16_t* Q  = qh + (size_t)(batch * HH + h) * SS * DK;
  const bf16_t* K  = kh + (size_t)(batch * HH + h) * SS * DK;
  const bf16_t* VT = vt + (size_t)(batch * HH + h) * DK * SS;  // [DK][SS]

  int nn = lane & 15;
  int mb = (lane & 16) ? 8 : 0;

  // -------- phase 1: scores = Q_tile @ K^T * (1/sqrt(64)) into LDS
  v16bf a0 = frag_a(Q + (size_t)q0 * DK + 0,  DK, lane);
  v16bf a1 = frag_a(Q + (size_t)q0 * DK + 32, DK, lane);
  for (int ct = wave; ct < SS / 16; ct += 4) {
    int n0 = ct * 16;
    const bf16_t* kp = K + (size_t)(n0 + nn) * DK + ((lane & 16) ? 16 : 0);
    v16bf b0 = *(const v16bf*)(kp);
    v16bf b1 = *(const v16bf*)(kp + 32);
    v8f acc = {};
    acc = wmma_bf16(a0, b0, acc);
    acc = wmma_bf16(a1, b1, acc);
#pragma unroll
    for (int v = 0; v < 8; ++v)
      sc[(size_t)(mb + v) * SCLD + n0 + nn] = acc[v] * 0.125f;
  }
  __syncthreads();

  // -------- phase 2: softmax over each of 16 rows (8 threads per row, lane-interleaved)
  int r = tid >> 3, j = tid & 7;
  float*  row  = sc + (size_t)r * SCLD;
  bf16_t* prow = pb + (size_t)r * PLD;
  float mx = -3.402823466e38f;
  for (int i = j; i < SS; i += 8) mx = fmaxf(mx, row[i]);
  red[tid] = mx;
  __syncthreads();
  float rmx = red[r * 8];
#pragma unroll
  for (int i = 1; i < 8; ++i) rmx = fmaxf(rmx, red[r * 8 + i]);
  __syncthreads();
  float ps = 0.f;
  for (int i = j; i < SS; i += 8) { float e = __expf(row[i] - rmx); row[i] = e; ps += e; }
  red[tid] = ps;
  __syncthreads();
  float sum = red[r * 8];
#pragma unroll
  for (int i = 1; i < 8; ++i) sum += red[r * 8 + i];
  float inv = 1.0f / sum;
  float* arow = attn + ((size_t)(batch * HH + h) * SS + (q0 + r)) * SS;
  for (int i = j; i < SS; i += 8) {
    float p = row[i] * inv;
    arow[i] = p;                 // fp32 attention weights (required output)
    prow[i] = (bf16_t)p;         // packed bf16 P for the P@V WMMAs
  }
  __syncthreads();

  // -------- phase 3: context = P(16x2048, LDS bf16) @ V(2048x64 via V^T)
  int d0 = wave * 16;
  int kb = (lane & 16) ? 8 : 0;
  int khalf = (lane & 16) ? 16 : 0;
  const bf16_t* vrow = VT + (size_t)(d0 + nn) * SS + khalf;  // B col d = d0+nn
  const bf16_t* parow = pb + (size_t)nn * PLD + khalf;       // A row m = lane&15
  v8f cacc = {};
  for (int k0 = 0; k0 < SS; k0 += 32) {
    v16bf a = pack16(*(const v8bf*)(parow + k0), *(const v8bf*)(parow + k0 + 8));
    v16bf b = *(const v16bf*)(vrow + k0);
    cacc = wmma_bf16(a, b, cacc);
  }
  bf16_t* cp = ctx + (size_t)(batch * SS + q0 + mb) * DD + h * DK + d0 + nn;
#pragma unroll
  for (int v = 0; v < 8; ++v) cp[(size_t)v * DD] = (bf16_t)cacc[v];
}

// ---------------------------------------------------------------- output projection
__global__ void k_out(const bf16_t* __restrict__ X, const bf16_t* __restrict__ W,
                      const float* __restrict__ bias, float* __restrict__ out) {
  int lane = threadIdx.x & 31;
  int waveId = blockIdx.x * (blockDim.x >> 5) + (threadIdx.x >> 5);
  const int NT = DD / 16;
  int tm = waveId / NT, tn = waveId % NT;
  int m0 = tm * 16, n0 = tn * 16;
  v8f acc = {};
  for (int k0 = 0; k0 < DD; k0 += 32) {
    v16bf a = frag_a(X + (size_t)m0 * DD + k0, DD, lane);
    v16bf b = frag_b_rows(W + (size_t)n0 * DD + k0, DD, lane);
    acc = wmma_bf16(a, b, acc);
  }
  int n = n0 + (lane & 15);
  float bia = bias[n];
  int mbase = m0 + ((lane & 16) ? 8 : 0);
#pragma unroll
  for (int v = 0; v < 8; ++v) {
    int rr = mbase + v;
    out[(size_t)rr * DD + n] = acc[v] + bia;
  }
}

extern "C" void kernel_launch(void* const* d_in, const int* in_sizes, int n_in,
                              void* d_out, int out_size, void* d_ws, size_t ws_size,
                              hipStream_t stream) {
  (void)in_sizes; (void)n_in; (void)out_size; (void)ws_size;
  const float* query = (const float*)d_in[0];
  const float* key_i = (const float*)d_in[1];
  const float* value = (const float*)d_in[2];
  const float* Wq = (const float*)d_in[3];
  const float* bq = (const float*)d_in[4];
  const float* Wk = (const float*)d_in[5];
  const float* bk = (const float*)d_in[6];
  const float* Wv = (const float*)d_in[7];
  const float* bv = (const float*)d_in[8];
  const float* Wo = (const float*)d_in[9];
  const float* bo = (const float*)d_in[10];

  float* out  = (float*)d_out;                       // [B,S,768]
  float* attn = out + (size_t)BS * DD;               // [B,H,S,S]

  bf16_t* w = (bf16_t*)d_ws;
  size_t o = 0;
  bf16_t* xq  = w + o; o += (size_t)BS * DD;
  bf16_t* xk  = w + o; o += (size_t)BS * DD;
  bf16_t* xv  = w + o; o += (size_t)BS * DD;
  bf16_t* wqb = w + o; o += (size_t)DD * DD;
  bf16_t* wkb = w + o; o += (size_t)DD * DD;
  bf16_t* wvb = w + o; o += (size_t)DD * DD;
  bf16_t* wob = w + o; o += (size_t)DD * DD;
  bf16_t* qh  = w + o; o += (size_t)BS * DD;
  bf16_t* kh  = w + o; o += (size_t)BS * DD;
  bf16_t* vt  = w + o; o += (size_t)BS * DD;         // V^T, [B,H,DK,S]
  bf16_t* ctx = w + o; o += (size_t)BS * DD;

  const int ncvt = BS * DD;
  k_cvt<<<1024, 256, 0, stream>>>(query, xq, ncvt);
  k_cvt<<<1024, 256, 0, stream>>>(key_i, xk, ncvt);
  k_cvt<<<1024, 256, 0, stream>>>(value, xv, ncvt);
  k_cvt<<<512, 256, 0, stream>>>(Wq, wqb, DD * DD);
  k_cvt<<<512, 256, 0, stream>>>(Wk, wkb, DD * DD);
  k_cvt<<<512, 256, 0, stream>>>(Wv, wvb, DD * DD);
  k_cvt<<<512, 256, 0, stream>>>(Wo, wob, DD * DD);

  const int projBlocks = (BS / 16) * (DD / 16) / 4;  // 3072 blocks, 4 waves each
  k_proj <<<projBlocks, 128, 0, stream>>>(xq, wqb, bq, qh);
  k_proj <<<projBlocks, 128, 0, stream>>>(xk, wkb, bk, kh);
  k_projT<<<projBlocks, 128, 0, stream>>>(xv, wvb, bv, vt);

  const size_t shmem = (size_t)(16 * SCLD + 128) * sizeof(float)
                     + (size_t)(16 * PLD) * sizeof(bf16_t);   // ~198.9 KB < 320 KB WGP LDS
  k_attn<<<BB * HH * (SS / 16), 128, shmem, stream>>>(qh, kh, vt, attn, ctx);

  k_out<<<projBlocks, 128, 0, stream>>>(ctx, wob, bo, out);
}
